// CNNLSTM_55327768708455
// MI455X (gfx1250) — matrix operand order repaired and checked by reading, not compiled, not run on hardware
//
#include <hip/hip_runtime.h>
#include <math.h>

typedef __attribute__((ext_vector_type(16))) _Float16 v16h;
typedef __attribute__((ext_vector_type(8)))  _Float16 v8h;
typedef __attribute__((ext_vector_type(8)))  float    v8f;
typedef __attribute__((ext_vector_type(4)))  float    v4f;

#define WMMA16(a, b, c) \
    __builtin_amdgcn_wmma_f32_16x16x32_f16(false, (a), false, (b), (short)0, (c), false, false)

#define N_T   512     // conv batch == LSTM time
#define LSEQ  1024    // conv length == LSTM batch
#define HD    64
#define EPSB  1e-5f
#define ZST   260     // zbuf column stride (floats): 4-bank skew per column

// ---------------------------------------------------------------------------
// Fragment loaders (CDNA5 wave32 WMMA 16x16x32 f16; contiguous k = 16*(lane>>4)+s in A and B)
// ---------------------------------------------------------------------------

// A-matrix 16x32 f16 from row-major [M][ld] (K contiguous).
// lane holds row m0+(lane&15); K = {a..a+16}, a = 16*(lane>>4).
__device__ __forceinline__ v16h load_a16(const _Float16* W, int ld, int m0, int k0, int lane) {
    int m = m0 + (lane & 15);
    int a = (lane >> 4) * 16;
    const _Float16* p = W + (size_t)m * ld + k0 + a;
    v8h lo = *(const v8h*)p;
    v8h hi = *(const v8h*)(p + 8);
    v16h r;
#pragma unroll
    for (int i = 0; i < 8; ++i) { r[i] = lo[i]; r[i + 8] = hi[i]; }
    return r;
}

// fast sigmoid / tanh on the sequential LSTM path (v_exp_f32 + v_rcp_f32)
__device__ __forceinline__ float sigm(float x) {
    return __builtin_amdgcn_rcpf(1.0f + __expf(-x));
}
__device__ __forceinline__ float tanh_fast(float x) {
    return fmaf(2.0f, sigm(2.0f * x), -1.0f);
}

// ---------------------------------------------------------------------------
// Prep kernels: fold BN into per-channel affine, convert weights to f16
// ---------------------------------------------------------------------------
__global__ void k_prep_bn(const float* b1, const float* g1, const float* be1, const float* m1, const float* v1,
                          const float* b2, const float* g2, const float* be2, const float* m2, const float* v2,
                          const float* b3, const float* g3, const float* be3, const float* m3, const float* v3,
                          const float* b4, float* scaleO, float* biasO) {
    int tid = threadIdx.x;
    int stage = tid >> 6, c = tid & 63;
    float s, b;
    if (stage == 0) { s = g1[c] * rsqrtf(v1[c] + EPSB); b = (b1[c] - m1[c]) * s + be1[c]; }
    else if (stage == 1) { s = g2[c] * rsqrtf(v2[c] + EPSB); b = (b2[c] - m2[c]) * s + be2[c]; }
    else if (stage == 2) { s = g3[c] * rsqrtf(v3[c] + EPSB); b = (b3[c] - m3[c]) * s + be3[c]; }
    else { s = 1.0f; b = b4[c]; }
    scaleO[stage * 64 + c] = s;
    biasO[stage * 64 + c]  = b;
}

// conv:  W0[co][ci] = w[co][ci][0], W1 = w[co][ci][1]   (w: [64][64][2])
// convT: W0[co][ci] = w[ci][co][1], W1 = w[ci][co][0]   (flip+transpose)
__global__ void k_prep_wpair(const float* src, _Float16* W0, _Float16* W1, int transpose) {
    int idx = blockIdx.x * 256 + threadIdx.x;
    if (idx >= 64 * 64) return;
    int co = idx >> 6, ci = idx & 63;
    if (!transpose) {
        W0[idx] = (_Float16)src[(co * 64 + ci) * 2 + 0];
        W1[idx] = (_Float16)src[(co * 64 + ci) * 2 + 1];
    } else {
        W0[idx] = (_Float16)src[(ci * 64 + co) * 2 + 1];
        W1[idx] = (_Float16)src[(ci * 64 + co) * 2 + 0];
    }
}

__global__ void k_prep_lstm(const float* wih, const float* whh, const float* bih, const float* bhh,
                            _Float16* Wih, _Float16* Whh, float* biasL) {
    int idx = blockIdx.x * 256 + threadIdx.x;
    if (idx < 256 * 64) {
        Wih[idx] = (_Float16)wih[idx];
        Whh[idx] = (_Float16)whh[idx];
    }
    if (idx < 256) biasL[idx] = bih[idx] + bhh[idx];
}

// ---------------------------------------------------------------------------
// conv1: Cin=6, K=2, valid. Memory-bound scalar kernel. in fp32 [n][1024][6]
// out f16 [n][1023][64]
// ---------------------------------------------------------------------------
__global__ void k_conv1(const float* __restrict__ X, _Float16* __restrict__ Y,
                        const float* __restrict__ w, const float* __restrict__ scale,
                        const float* __restrict__ bias) {
    int idx = blockIdx.x * 256 + threadIdx.x;
    int co = idx & 63;
    int t  = (idx >> 6) % 1023;
    int n  = idx / (1023 * 64);
    if (n >= N_T) return;
    const float* xp = X + ((size_t)n * LSEQ + t) * 6;
    float acc = 0.0f;
#pragma unroll
    for (int k = 0; k < 2; ++k)
#pragma unroll
        for (int ci = 0; ci < 6; ++ci)
            acc += w[(co * 6 + ci) * 2 + k] * xp[k * 6 + ci];
    float v = fmaxf(acc * scale[co] + bias[co], 0.0f);
    Y[((size_t)n * 1023 + t) * HD + co] = (_Float16)v;
}

// ---------------------------------------------------------------------------
// Conv/ConvT as dual 64x64 GEMM with shifts s0,s1 + affine + ReLU, WMMA f16.
// X: f16 [n][Lin][64], Y: f16 [n][Lout][64].
// Block = 256 (8 waves): 64 co x 256 t-cols. Wave -> 1 co-tile x 8 t-tiles.
// Interior tiles take an unpredicated fast path; edges zero-pad per lane.
// ---------------------------------------------------------------------------
__global__ void __launch_bounds__(256) k_conv_gemm(const _Float16* __restrict__ X,
                                                   _Float16* __restrict__ Y,
                                                   const _Float16* __restrict__ W0,
                                                   const _Float16* __restrict__ W1,
                                                   const float* __restrict__ scale,
                                                   const float* __restrict__ bias,
                                                   int Lin, int Lout, int s0, int s1) {
    int n     = blockIdx.y;
    int tbase = blockIdx.x * 256;
    int wave  = threadIdx.x >> 5;
    int lane  = threadIdx.x & 31;
    int m0    = (wave >> 1) * 16;  // co tile

    const _Float16* Xn = X + (size_t)n * Lin * HD;
    _Float16*       Yn = Y + (size_t)n * Lout * HD;

    // loop-invariant A fragments: 2 taps x 2 K-chunks
    v16h a00 = load_a16(W0, 64, m0, 0, lane);
    v16h a01 = load_a16(W0, 64, m0, 32, lane);
    v16h a10 = load_a16(W1, 64, m0, 0, lane);
    v16h a11 = load_a16(W1, 64, m0, 32, lane);

    int ncol = (lane & 15), koff = (lane >> 4) * 16;
    int mb   = m0 + 8 * (lane >> 4);   // 8 contiguous output channels per lane

    float sc[8], bi[8];
#pragma unroll
    for (int r = 0; r < 8; ++r) { sc[r] = scale[mb + r]; bi[r] = bias[mb + r]; }

#pragma unroll 2
    for (int i = 0; i < 8; ++i) {
        int t0 = tbase + ((wave & 1) * 8 + i) * 16;
        if (t0 >= Lout) continue;  // wave-uniform

        bool interior = (t0 + s0 >= 0) && (t0 + 15 + s1 < Lin) && (t0 + 15 < Lout);

        v16h b00, b01, b10, b11;
        if (interior) {
            const _Float16* xb = Xn + (size_t)(t0 + ncol) * HD + koff;
            b00 = *(const v16h*)(xb + s0 * HD);
            b01 = *(const v16h*)(xb + s0 * HD + 32);
            b10 = *(const v16h*)(xb + s1 * HD);
            b11 = *(const v16h*)(xb + s1 * HD + 32);
        } else {
            v16h bz = {};
            int c0 = t0 + ncol + s0;
            int c1 = t0 + ncol + s1;
            b00 = (c0 >= 0 && c0 < Lin) ? *(const v16h*)(Xn + (size_t)c0 * HD + koff)      : bz;
            b01 = (c0 >= 0 && c0 < Lin) ? *(const v16h*)(Xn + (size_t)c0 * HD + 32 + koff) : bz;
            b10 = (c1 >= 0 && c1 < Lin) ? *(const v16h*)(Xn + (size_t)c1 * HD + koff)      : bz;
            b11 = (c1 >= 0 && c1 < Lin) ? *(const v16h*)(Xn + (size_t)c1 * HD + 32 + koff) : bz;
        }

        v8f acc = {};
        acc = WMMA16(a00, b00, acc);
        acc = WMMA16(a01, b01, acc);
        acc = WMMA16(a10, b10, acc);
        acc = WMMA16(a11, b11, acc);

        // epilogue: affine + ReLU + cvt, one b128 store (8 contiguous channels)
        int tc = t0 + ncol;
        if (tc < Lout) {
            v8h o;
#pragma unroll
            for (int r = 0; r < 8; ++r)
                o[r] = (_Float16)fmaxf(acc[r] * sc[r] + bi[r], 0.0f);
            *(v8h*)(Yn + (size_t)tc * HD + mb) = o;
        }
    }
}

// ---------------------------------------------------------------------------
// Persistent LSTM: 64 blocks x 16 columns; h/c resident in LDS for all 512
// steps. 16 waves; wave w owns gate rows [16w,16w+16) x the block's 16 cols.
// Per step, per wave: 4x v_wmma_f32_16x16x32_f16 (Wih and Whh, K=64 each).
// Output projection: wave w == column w, shfl_xor wave32 reduction.
// ---------------------------------------------------------------------------
__global__ void __launch_bounds__(512) k_lstm(const _Float16* __restrict__ Xs,
                                              const _Float16* __restrict__ Wih,
                                              const _Float16* __restrict__ Whh,
                                              const float* __restrict__ biasL,
                                              const float* __restrict__ h0,
                                              const float* __restrict__ c0,
                                              const float* __restrict__ ow,
                                              const float* __restrict__ ob,
                                              float* __restrict__ out) {
    __shared__ __align__(16) float    zbuf[16 * ZST];   // [col][gate], skewed
    __shared__ __align__(16) float    hS[16 * 64];      // [col][hid] fp32
    __shared__ __align__(16) float    cS[16 * 64];
    __shared__ __align__(32) _Float16 hB[16 * 64];      // [col][hid] f16 (B-frag source)

    int col0 = blockIdx.x * 16;
    int wave = threadIdx.x >> 5;
    int lane = threadIdx.x & 31;
    int m0   = wave * 16;

    for (int i = threadIdx.x; i < 16 * 64; i += 512) {
        int col = i >> 6, hid = i & 63;
        float hv = h0[(size_t)(col0 + col) * 64 + hid];
        hS[i] = hv;
        cS[i] = c0[(size_t)(col0 + col) * 64 + hid];
        hB[i] = (_Float16)hv;
    }

    // loop-invariant weight A-fragments + bias accumulator seed
    v16h aih0 = load_a16(Wih, 64, m0, 0, lane);
    v16h aih1 = load_a16(Wih, 64, m0, 32, lane);
    v16h ahh0 = load_a16(Whh, 64, m0, 0, lane);
    v16h ahh1 = load_a16(Whh, 64, m0, 32, lane);

    int bn = (lane & 15), bk = (lane >> 4) * 16;
    int mb = m0 + 8 * (lane >> 4);       // 8 contiguous gate rows per lane
    int cw = lane & 15;                  // column this lane's D elems belong to

    v8f cb;
#pragma unroll
    for (int r = 0; r < 8; ++r) cb[r] = biasL[mb + r];

    // per-lane output-projection weights (wave == column)
    float owa = ow[2 * lane], owb = ow[2 * lane + 1];
    float ob0 = ob[0];

    __syncthreads();

    for (int t = 0; t < N_T; ++t) {
        const _Float16* xt = Xs + ((size_t)t * LSEQ + col0) * HD;

        v16h xb0 = *(const v16h*)(xt + bn * HD + bk);
        v16h xb1 = *(const v16h*)(xt + bn * HD + 32 + bk);
        v16h hb0 = *(const v16h*)(hB + bn * HD + bk);
        v16h hb1 = *(const v16h*)(hB + bn * HD + 32 + bk);

        // prefetch next step's x tile (global_prefetch_b8)
        if (t + 1 < N_T)
            __builtin_prefetch(xt + (size_t)LSEQ * HD + bn * HD + bk, 0, 0);

        v8f acc = cb;
        acc = WMMA16(aih0, xb0, acc);
        acc = WMMA16(aih1, xb1, acc);
        acc = WMMA16(ahh0, hb0, acc);
        acc = WMMA16(ahh1, hb1, acc);

        // store D tile: two b128 LDS writes, bank-skewed [col][gate] layout
        {
            v4f lo, hi;
#pragma unroll
            for (int r = 0; r < 4; ++r) { lo[r] = acc[r]; hi[r] = acc[r + 4]; }
            float* zp = &zbuf[cw * ZST + mb];
            *(v4f*)zp       = lo;
            *(v4f*)(zp + 4) = hi;
        }

        __syncthreads();

        for (int i = threadIdx.x; i < 16 * 64; i += 512) {
            int col = i >> 6, hid = i & 63;
            const float* z = &zbuf[col * ZST + hid];
            float zi = z[0], zf = z[64], zg = z[128], zo = z[192];
            float cn = sigm(zf) * cS[i] + sigm(zi) * tanh_fast(zg);
            float hn = sigm(zo) * tanh_fast(cn);
            cS[i] = cn;
            hS[i] = hn;
            hB[i] = (_Float16)hn;
        }

        __syncthreads();

        // output projection: wave w handles column w; 2 hidden elems per lane
        {
            const float* hp = &hS[wave * 64 + 2 * lane];
            float part = owa * hp[0] + owb * hp[1];
#pragma unroll
            for (int off = 16; off > 0; off >>= 1)
                part += __shfl_xor(part, off, 32);
            if (lane == 0)
                out[(size_t)t * LSEQ + col0 + wave] = part + ob0;
        }
    }

    // final hT, cT: (1,1024,64) each, appended after outputs (512*1024)
    const size_t offH = (size_t)N_T * LSEQ;
    const size_t offC = offH + (size_t)LSEQ * 64;
    for (int i = threadIdx.x; i < 16 * 64; i += 512) {
        int col = i >> 6, hid = i & 63;
        out[offH + (size_t)(col0 + col) * 64 + hid] = hS[i];
        out[offC + (size_t)(col0 + col) * 64 + hid] = cS[i];
    }
}

// ---------------------------------------------------------------------------
extern "C" void kernel_launch(void* const* d_in, const int* in_sizes, int n_in,
                              void* d_out, int out_size, void* d_ws, size_t ws_size,
                              hipStream_t stream) {
    (void)in_sizes; (void)n_in; (void)out_size; (void)ws_size;

    size_t off = 0;
    auto alloc = [&](size_t bytes) -> void* {
        off = (off + 255) & ~(size_t)255;
        void* r = (char*)d_ws + off;
        off += bytes;
        return r;
    };

    const size_t actElems = (size_t)N_T * LSEQ * HD;
    _Float16* actA = (_Float16*)alloc(actElems * sizeof(_Float16));
    _Float16* actB = (_Float16*)alloc(actElems * sizeof(_Float16));
    float*  scaleB = (float*)alloc(4 * 64 * sizeof(float));
    float*  biasB  = (float*)alloc(4 * 64 * sizeof(float));
    _Float16* W2_0 = (_Float16*)alloc(4096 * sizeof(_Float16));
    _Float16* W2_1 = (_Float16*)alloc(4096 * sizeof(_Float16));
    _Float16* W3_0 = (_Float16*)alloc(4096 * sizeof(_Float16));
    _Float16* W3_1 = (_Float16*)alloc(4096 * sizeof(_Float16));
    _Float16* W4_0 = (_Float16*)alloc(4096 * sizeof(_Float16));
    _Float16* W4_1 = (_Float16*)alloc(4096 * sizeof(_Float16));
    _Float16* Wih  = (_Float16*)alloc(256 * 64 * sizeof(_Float16));
    _Float16* Whh  = (_Float16*)alloc(256 * 64 * sizeof(_Float16));
    float*  biasL  = (float*)alloc(256 * sizeof(float));

#define IN(i) ((const float*)d_in[(i)])

    k_prep_bn<<<1, 256, 0, stream>>>(IN(4), IN(5), IN(6), IN(7), IN(8),
                                     IN(10), IN(11), IN(12), IN(13), IN(14),
                                     IN(16), IN(17), IN(18), IN(19), IN(20),
                                     IN(22), scaleB, biasB);
    k_prep_wpair<<<16, 256, 0, stream>>>(IN(9),  W2_0, W2_1, 0);
    k_prep_wpair<<<16, 256, 0, stream>>>(IN(15), W3_0, W3_1, 1);
    k_prep_wpair<<<16, 256, 0, stream>>>(IN(21), W4_0, W4_1, 1);
    k_prep_lstm<<<64, 256, 0, stream>>>(IN(23), IN(24), IN(25), IN(26), Wih, Whh, biasL);

    // conv1: (512,1024,6) -> actA (512,1023,64)
    {
        long total = (long)N_T * 1023 * 64;
        k_conv1<<<(int)((total + 255) / 256), 256, 0, stream>>>(IN(0), actA, IN(3), scaleB, biasB);
    }
    // conv2: actA(1023) -> actB(1022), valid (s=0,1)
    k_conv_gemm<<<dim3(4, N_T), 256, 0, stream>>>(actA, actB, W2_0, W2_1,
                                                  scaleB + 64, biasB + 64, 1023, 1022, 0, 1);
    // tc1: actB(1022) -> actA(1023), convT pad 1 (s=-1,0)
    k_conv_gemm<<<dim3(4, N_T), 256, 0, stream>>>(actB, actA, W3_0, W3_1,
                                                  scaleB + 128, biasB + 128, 1022, 1023, -1, 0);
    // tc2: actA(1023) -> actB(1024), convT pad 1, ReLU only
    k_conv_gemm<<<dim3(4, N_T), 256, 0, stream>>>(actA, actB, W4_0, W4_1,
                                                  scaleB + 192, biasB + 192, 1023, 1024, -1, 0);
    // LSTM over T=512, batch columns 1024, + output projection + hT/cT
    k_lstm<<<64, 512, 0, stream>>>(actB, Wih, Whh, biasL, IN(1), IN(2), IN(27), IN(28),
                                   (float*)d_out);
#undef IN
}
